// HumanVAttention_79826262163620
// MI455X (gfx1250) — matrix operand order repaired and checked by reading, not compiled, not run on hardware
//
#include <hip/hip_runtime.h>

typedef __attribute__((ext_vector_type(16))) __bf16        v16bf;
typedef __attribute__((ext_vector_type(8)))  __bf16        v8bf;
typedef __attribute__((ext_vector_type(8)))  float         v8f;
typedef __attribute__((ext_vector_type(4)))  unsigned int  u32x4;

constexpr int CB = 2, CS = 4096, CHID = 2048, CH = 16, CKV = 4, CD = 128;
constexpr int CBS = 64, CL = 4, CG = 4, CSTRIDE = 4, CWTOK = 2048;
constexpr int CNB = CS / CBS;          // 64 query blocks
constexpr int CREP = CH / CKV;         // 4

// ---------------- WMMA fragment helpers (CDNA5 ISA 7.12.2 layouts) ----------

__device__ __forceinline__ v16bf make_frag(u32x4 lo, u32x4 hi) {
  union { v16bf v; u32x4 q[2]; } u;
  u.q[0] = lo; u.q[1] = hi;
  return u.v;
}

// A-matrix 16x32 bf16, row-major source. `row` points to A[m][kblock*32].
// lanes 0-15: VGPR0-3 K=0..7, VGPR4-7 K=16..23 ; lanes 16-31: K=8..15, 24..31
__device__ __forceinline__ v16bf load_a_frag(const __bf16* row, int lane) {
  const int kh = (lane < 16) ? 0 : 8;
  u32x4 lo = *(const u32x4*)(row + kh);
  u32x4 hi = *(const u32x4*)(row + kh + 16);
  return make_frag(lo, hi);
}

// B-matrix 32x16 bf16 from N-major (BT[n][k]) source. `row` -> BT[n][kblock*32].
// lanes 0-15: K=0..7, 16..23 ; lanes 16-31: K=8..15, 24..31 (matches A).
__device__ __forceinline__ v16bf load_b_frag(const __bf16* row, int lane) {
  const int kh = (lane < 16) ? 0 : 8;
  u32x4 lo = *(const u32x4*)(row + kh);
  u32x4 hi = *(const u32x4*)(row + kh + 16);
  return make_frag(lo, hi);
}

// ------- bf16 WMMA GEMM, 64x64 macro-tile per wave (4x4 register blocking) --
// C[M,N] = A[M,K] * BT[N,K]^T ; M,N multiples of 64, K multiple of 32.

__global__ __launch_bounds__(256) void gemm_bf16(const __bf16* __restrict__ A,
                                                 const __bf16* __restrict__ BT,
                                                 float* __restrict__ C,
                                                 int M, int N, int K) {
  const int lane = threadIdx.x & 31;
  const int wave = blockIdx.x * 8 + (threadIdx.x >> 5);
  const int tilesN = N >> 6;
  const int tilesM = M >> 6;
  if (wave >= tilesM * tilesN) return;
  const int tm = (wave / tilesN) << 6;
  const int tn = (wave % tilesN) << 6;
  const int l15 = lane & 15;

  const __bf16* arow[4];
  const __bf16* brow[4];
#pragma unroll
  for (int mt = 0; mt < 4; ++mt) arow[mt] = A  + (long)(tm + mt * 16 + l15) * K;
#pragma unroll
  for (int nt = 0; nt < 4; ++nt) brow[nt] = BT + (long)(tn + nt * 16 + l15) * K;

  v8f acc[4][4] = {};
  for (int k = 0; k < K; k += 32) {
    v16bf a[4], b[4];
#pragma unroll
    for (int mt = 0; mt < 4; ++mt) a[mt] = load_a_frag(arow[mt] + k, lane);
#pragma unroll
    for (int nt = 0; nt < 4; ++nt) b[nt] = load_b_frag(brow[nt] + k, lane);
#pragma unroll
    for (int mt = 0; mt < 4; ++mt)
#pragma unroll
      for (int nt = 0; nt < 4; ++nt)
        acc[mt][nt] = __builtin_amdgcn_wmma_f32_16x16x32_bf16(
            false, a[mt], false, b[nt], (short)0, acc[mt][nt], false, false);
  }

#pragma unroll
  for (int mt = 0; mt < 4; ++mt) {
    const int rb = tm + mt * 16 + ((lane < 16) ? 0 : 8);
#pragma unroll
    for (int nt = 0; nt < 4; ++nt) {
      const int col = tn + nt * 16 + l15;
#pragma unroll
      for (int v = 0; v < 8; ++v)
        C[(long)(rb + v) * N + col] = acc[mt][nt][v];
    }
  }
}

// ---------------- data-prep kernels -----------------------------------------

__global__ void cvt_bf16(const float* __restrict__ src, __bf16* __restrict__ dst,
                         long n) {
  long i = (long)blockIdx.x * blockDim.x + threadIdx.x;
  if (i < n) dst[i] = (__bf16)src[i];
}

// src[K][N] f32 -> dst[N][K] bf16, LDS-tiled 32x32 (coalesced both sides).
// K, N multiples of 32.
__global__ __launch_bounds__(256) void transpose_bf16(
    const float* __restrict__ src, __bf16* __restrict__ dst, int K, int N) {
  __shared__ __bf16 tile[32][33];
  const int tilesK = K >> 5;
  const int tk = (int)(blockIdx.x % tilesK);
  const int tn = (int)(blockIdx.x / tilesK);
  const int c  = threadIdx.x & 31;
  const int r0 = (threadIdx.x >> 5) * 4;
#pragma unroll
  for (int i = 0; i < 4; ++i) {
    const int k = tk * 32 + r0 + i;
    const int n = tn * 32 + c;
    tile[r0 + i][c] = (__bf16)src[(long)k * N + n];
  }
  __syncthreads();
#pragma unroll
  for (int i = 0; i < 4; ++i) {
    const int n = tn * 32 + r0 + i;
    const int k = tk * 32 + c;
    dst[(long)n * K + k] = tile[c][r0 + i];
  }
}

// xf[B*S][nh*D] f32 --rope--> xout[B][nh][S][D] bf16
__global__ void rope_pack(const float* __restrict__ xf,
                          const float* __restrict__ cosb,
                          const float* __restrict__ sinb,
                          __bf16* __restrict__ xout, int nh) {
  long i = (long)blockIdx.x * blockDim.x + threadIdx.x;
  long total = (long)CB * nh * CS * CD;
  if (i >= total) return;
  int d = (int)(i & (CD - 1));
  long t = i >> 7;                       // CD == 128
  int s = (int)(t % CS); t /= CS;
  int h = (int)(t % nh);
  int b = (int)(t / nh);
  long src = ((long)b * CS + s) * ((long)nh * CD) + (long)h * CD + d;
  float x  = xf[src];
  float xr = (d < CD / 2) ? -xf[src + CD / 2] : xf[src - CD / 2];
  long cs = ((long)b * CS + s) * CD + d;
  xout[i] = (__bf16)(x * cosb[cs] + xr * sinb[cs]);
}

// vf[B*S][KV*D] f32 -> vtb[B][KV][D][S] bf16 (transposed for PV B-fragments)
__global__ void v_pack_t(const float* __restrict__ vf, __bf16* __restrict__ vtb) {
  long i = (long)blockIdx.x * blockDim.x + threadIdx.x;
  long total = (long)CB * CKV * CD * CS;
  if (i >= total) return;
  int s = (int)(i % CS);
  long t = i / CS;
  int d = (int)(t % CD); t /= CD;
  int kvh = (int)(t % CKV);
  int b = (int)(t / CKV);
  long src = ((long)b * CS + s) * (CKV * CD) + (long)kvh * CD + d;
  vtb[i] = (__bf16)vf[src];
}

// ---------------- block-sparse attention ------------------------------------
// 1 workgroup per (b,h,qblock); 8 waves, wave w owns key-block w
// (w<4: local blocks qblk-3..qblk ; w>=4: strided global blocks)

__global__ __launch_bounds__(256) void attn_wmma(
    const __bf16* __restrict__ qb, const __bf16* __restrict__ kb,
    const __bf16* __restrict__ vt, const float* __restrict__ mask2d,
    __bf16* __restrict__ attn) {
  __shared__ __bf16 sQ[CBS * CD];        // 64 x 128 Q block (16 KB)
  __shared__ __bf16 sP[CBS * 8 * CBS];   // 64 x 512 scores/probs (64 KB)
  __shared__ float  sInv[CBS];

  const int bid  = blockIdx.x;
  const int qblk = bid % CNB;
  const int hh   = (bid / CNB) % CH;
  const int bb   = bid / (CNB * CH);
  const int kv   = hh / CREP;
  const int lane = threadIdx.x & 31;
  const int wv   = threadIdx.x >> 5;
  const int l15  = lane & 15;

  // key-block indices + validity (matches reference _global_idx)
  int kbi[8]; float kbv[8];
#pragma unroll
  for (int w = 0; w < CL; ++w) {
    int ib = qblk - (CL - 1) + w;
    kbv[w] = (ib >= 0) ? 1.f : 0.f;
    kbi[w] = (ib >= 0) ? ib : 0;
  }
#pragma unroll
  for (int j = 0; j < CG; ++j) { kbi[CL + j] = 0; kbv[CL + j] = 0.f; }
  {
    int i = qblk;
    int maxaway = CWTOK / CBS; if (maxaway < 1) maxaway = 1;   // 32
    if (i >= 1) {
      int start = i - maxaway; if (start < 0) start = 0;
      int cnt  = (i - start + CSTRIDE - 1) / CSTRIDE;          // |range(start,i,4)|
      int take = cnt < CG ? cnt : CG;
      int s0   = CG - take;
      for (int t = 0; t < take; ++t) {
        kbi[CL + s0 + t] = start + CSTRIDE * (cnt - take + t);
        kbv[CL + s0 + t] = 1.f;
      }
    }
  }

  const int  myblk   = kbi[wv];
  const bool myvalid = kbv[wv] > 0.5f;
  const __bf16* qbase = qb + (((long)bb * CH + hh) * CS + (long)qblk * CBS) * CD;
  const __bf16* kbase = kb + (((long)bb * CKV + kv) * CS) * CD;

  // ---- stage Q block into LDS once (cooperative, coalesced b128) ----
  {
    const u32x4* srcq = (const u32x4*)qbase;        // 1024 x 16B chunks
    u32x4* dstq = (u32x4*)sQ;
#pragma unroll
    for (int i = 0; i < 4; ++i)
      dstq[threadIdx.x * 4 + i] = srcq[threadIdx.x * 4 + i];
  }
  __syncthreads();

  const float scale = 0.08838834764831845f;   // 128^-0.5
  const float neg   = -1.0e9f;

  // ---- scores: Q(64x128) x K^T(128x64); 4x4 register-blocked WMMA ----
  v8f acc[4][4] = {};
  if (myvalid) {
    for (int k = 0; k < CD; k += 32) {
      v16bf a[4], b[4];
#pragma unroll
      for (int mt = 0; mt < 4; ++mt)
        a[mt] = load_a_frag(&sQ[(mt * 16 + l15) * CD + k], lane);
#pragma unroll
      for (int nt = 0; nt < 4; ++nt)
        b[nt] = load_b_frag(kbase + ((long)myblk * CBS + nt * 16 + l15) * CD + k,
                            lane);
#pragma unroll
      for (int mt = 0; mt < 4; ++mt)
#pragma unroll
        for (int nt = 0; nt < 4; ++nt)
          acc[mt][nt] = __builtin_amdgcn_wmma_f32_16x16x32_bf16(
              false, a[mt], false, b[nt], (short)0, acc[mt][nt], false, false);
    }
  }
#pragma unroll
  for (int mt = 0; mt < 4; ++mt) {
#pragma unroll
    for (int nt = 0; nt < 4; ++nt) {
      const int n = nt * 16 + l15;
      float km   = myvalid ? mask2d[(long)bb * CS + myblk * CBS + n] : 0.f;
      float madd = (1.f - km) * neg;
      const int rb = mt * 16 + ((lane < 16) ? 0 : 8);
#pragma unroll
      for (int v = 0; v < 8; ++v) {
        float sc = acc[mt][nt][v] * scale + madd;
        if (wv == CL - 1 && n > rb + v) sc += neg;   // intra-block causal
        sP[(rb + v) * (8 * CBS) + wv * CBS + n] = (__bf16)sc;
      }
    }
  }
  __syncthreads();

  // ---- softmax: 4 threads per row, v8bf-vectorized LDS, shfl reductions ----
  {
    const int r  = threadIdx.x >> 2;          // 0..63
    const int qt = threadIdx.x & 3;           // quarter of the 512 columns
    __bf16* row = &sP[r * (8 * CBS) + qt * 128];
    float mx = -3.0e38f;
#pragma unroll
    for (int c8 = 0; c8 < 16; ++c8) {
      v8bf vv = *(const v8bf*)&row[c8 * 8];
#pragma unroll
      for (int j = 0; j < 8; ++j) mx = fmaxf(mx, (float)vv[j]);
    }
    mx = fmaxf(mx, __shfl_xor(mx, 1, 32));    // reduce across the 4-lane group
    mx = fmaxf(mx, __shfl_xor(mx, 2, 32));
    float sum = 0.f;
#pragma unroll
    for (int c8 = 0; c8 < 16; ++c8) {
      v8bf vv = *(const v8bf*)&row[c8 * 8];
      v8bf ev;
#pragma unroll
      for (int j = 0; j < 8; ++j) {
        float e = __expf((float)vv[j] - mx);
        sum += e;
        ev[j] = (__bf16)e;
      }
      *(v8bf*)&row[c8 * 8] = ev;
    }
    sum += __shfl_xor(sum, 1, 32);
    sum += __shfl_xor(sum, 2, 32);
    if (qt == 0) sInv[r] = 1.f / sum;
  }
  __syncthreads();

  // ---- O = P(64x512) x V(512x128); wave wv owns d-columns wv*16..+15 ----
  // B-frag loaded once per K-step, reused across the 4 row tiles.
  const __bf16* vbase =
      vt + (((long)bb * CKV + kv) * CD + wv * 16 + l15) * (long)CS;
  v8f oacc[4] = {};
  for (int ks = 0; ks < 16; ++ks) {
    const int kblk = ks >> 1;
    if (!(kbv[kblk] > 0.5f)) continue;           // masked block -> P ~ 0
    const __bf16* brow = vbase + (long)kbi[kblk] * CBS + (ks & 1) * 32;
    v16bf b = load_b_frag(brow, lane);
#pragma unroll
    for (int mt = 0; mt < 4; ++mt) {
      v16bf a = load_a_frag(&sP[(mt * 16 + l15) * (8 * CBS) + ks * 32], lane);
      oacc[mt] = __builtin_amdgcn_wmma_f32_16x16x32_bf16(
          false, a, false, b, (short)0, oacc[mt], false, false);
    }
  }
#pragma unroll
  for (int mt = 0; mt < 4; ++mt) {
    const int rb = mt * 16 + ((lane < 16) ? 0 : 8);
    const int n  = wv * 16 + l15;
#pragma unroll
    for (int v = 0; v < 8; ++v) {
      float o = oacc[mt][v] * sInv[rb + v];
      attn[((long)(bb * CS + qblk * CBS + rb + v)) * (CH * CD) + hh * CD + n] =
          (__bf16)o;
    }
  }
}

// ---------------- host driver -----------------------------------------------

extern "C" void kernel_launch(void* const* d_in, const int* in_sizes, int n_in,
                              void* d_out, int out_size, void* d_ws, size_t ws_size,
                              hipStream_t stream) {
  (void)in_sizes; (void)n_in; (void)out_size; (void)ws_size;
  const float* hs   = (const float*)d_in[0];
  const float* cosb = (const float*)d_in[1];
  const float* sinb = (const float*)d_in[2];
  const float* mask = (const float*)d_in[3];
  const float* Wq   = (const float*)d_in[4];
  const float* Wk   = (const float*)d_in[5];
  const float* Wv   = (const float*)d_in[6];
  const float* Wo   = (const float*)d_in[7];

  char* p = (char*)d_ws;
  auto take = [&](size_t bytes) {
    char* r = p; p += (bytes + 255) & ~(size_t)255; return r;
  };

  const long MB = (long)CB * CS;   // 8192 rows
  __bf16* hsb  = (__bf16*)take((size_t)MB * CHID * 2);
  __bf16* wqT  = (__bf16*)take((size_t)(CH * CD) * CHID * 2);
  __bf16* wkT  = (__bf16*)take((size_t)(CKV * CD) * CHID * 2);
  __bf16* wvT  = (__bf16*)take((size_t)(CKV * CD) * CHID * 2);
  __bf16* woT  = (__bf16*)take((size_t)CHID * (CH * CD) * 2);
  float*  qf   = (float*)take((size_t)MB * (CH * CD) * 4);
  float*  kf   = (float*)take((size_t)MB * (CKV * CD) * 4);
  float*  vf   = (float*)take((size_t)MB * (CKV * CD) * 4);
  __bf16* qpk  = (__bf16*)take((size_t)CB * CH * CS * CD * 2);
  __bf16* kpk  = (__bf16*)take((size_t)CB * CKV * CS * CD * 2);
  __bf16* vtb  = (__bf16*)take((size_t)CB * CKV * CD * CS * 2);
  __bf16* attn = (__bf16*)take((size_t)MB * (CH * CD) * 2);

  auto grid1 = [](long n) { return dim3((unsigned)((n + 255) / 256)); };
  auto gemmGrid = [](long M, long N) {
    long t = (M / 64) * (N / 64);          // one 64x64 macro-tile per wave
    return dim3((unsigned)((t + 7) / 8));
  };
  auto trGrid = [](long K, long N) { return dim3((unsigned)((K / 32) * (N / 32))); };

  // 1) precision conversion / weight transpose (LDS-tiled, coalesced)
  cvt_bf16<<<grid1(MB * CHID), 256, 0, stream>>>(hs, hsb, MB * CHID);
  transpose_bf16<<<trGrid(CHID, CH * CD), 256, 0, stream>>>(Wq, wqT, CHID, CH * CD);
  transpose_bf16<<<trGrid(CHID, CKV * CD), 256, 0, stream>>>(Wk, wkT, CHID, CKV * CD);
  transpose_bf16<<<trGrid(CHID, CKV * CD), 256, 0, stream>>>(Wv, wvT, CHID, CKV * CD);
  transpose_bf16<<<trGrid(CH * CD, CHID), 256, 0, stream>>>(Wo, woT, CH * CD, CHID);

  // 2) QKV projections (register-blocked WMMA)
  gemm_bf16<<<gemmGrid(MB, CH * CD), 256, 0, stream>>>(hsb, wqT, qf, (int)MB, CH * CD, CHID);
  gemm_bf16<<<gemmGrid(MB, CKV * CD), 256, 0, stream>>>(hsb, wkT, kf, (int)MB, CKV * CD, CHID);
  gemm_bf16<<<gemmGrid(MB, CKV * CD), 256, 0, stream>>>(hsb, wvT, vf, (int)MB, CKV * CD, CHID);

  // 3) RoPE + head-major repack (V transposed for PV)
  rope_pack<<<grid1((long)CB * CH * CS * CD), 256, 0, stream>>>(qf, cosb, sinb, qpk, CH);
  rope_pack<<<grid1((long)CB * CKV * CS * CD), 256, 0, stream>>>(kf, cosb, sinb, kpk, CKV);
  v_pack_t<<<grid1((long)CB * CKV * CD * CS), 256, 0, stream>>>(vf, vtb);

  // 4) block-sparse attention (WMMA scores + WMMA PV, Q staged in LDS)
  attn_wmma<<<dim3(CB * CH * CNB), 256, 0, stream>>>(qpk, kpk, vtb, mask, attn);

  // 5) output projection (register-blocked WMMA, f32 result)
  gemm_bf16<<<gemmGrid(MB, CHID), 256, 0, stream>>>(attn, woT, (float*)d_out,
                                                    (int)MB, CHID, CHID);
}
